// PseqStepV3_23338852287255
// MI455X (gfx1250) — matrix operand order repaired and checked by reading, not compiled, not run on hardware
//
#include <hip/hip_runtime.h>

#define N_NODES 50000
#define N_EDGES 800000
#define DCH     64

typedef __attribute__((ext_vector_type(16))) _Float16 v16h;
typedef __attribute__((ext_vector_type(8)))  _Float16 v8h;
typedef __attribute__((ext_vector_type(8)))  float    v8f;

// ---- monotone float<->uint encoding for atomicMax-based segment max ----
__device__ __forceinline__ unsigned enc_f32(float f) {
  unsigned u = __float_as_uint(f);
  return (u & 0x80000000u) ? ~u : (u | 0x80000000u);
}
__device__ __forceinline__ float dec_f32(unsigned u) {
  unsigned b = (u & 0x80000000u) ? (u & 0x7FFFFFFFu) : ~u;
  return __uint_as_float(b);
}

// ---- pre-swizzle one layer's 4 weight matrices (f32 [64][64]) into the
//      per-lane B layout this file uses (k = 16*half + j) for v_wmma_f32_16x16x32_f16 ----
// dest idx: (((mat*4 + ntile)*2 + kchunk)*32 + lane)*16 + j
// lane = (n&15) | (((k&31)>>4)<<4), j = k&15 within chunk
__global__ void GtcWconv_kernel(const float* __restrict__ Wq, const float* __restrict__ Wk,
                                const float* __restrict__ Wv, const float* __restrict__ Ws,
                                _Float16* __restrict__ dst) {
  int tid = blockIdx.x * blockDim.x + threadIdx.x;
  if (tid >= 4 * 64 * 64) return;
  int m = tid >> 12;
  int r = tid & 4095;
  int k = r >> 6;
  int n = r & 63;
  const float* W = (m == 0) ? Wq : (m == 1) ? Wk : (m == 2) ? Wv : Ws;
  float val = W[k * 64 + n];
  int t  = n >> 4;
  int c  = k >> 5;
  int kk = k & 31;
  int j  = kk & 15;
  int hi = kk >> 4;
  int lane = (n & 15) | (hi << 4);
  dst[(((m * 4 + t) * 2 + c) * 32 + lane) * 16 + j] = (_Float16)val;
}

__global__ void GtcF32toF16_kernel(const float* __restrict__ in, _Float16* __restrict__ out, int n) {
  int i = blockIdx.x * blockDim.x + threadIdx.x;
  if (i < n) out[i] = (_Float16)in[i];
}

__global__ void GtcInit_kernel(float* __restrict__ agg, float* __restrict__ denom,
                               unsigned* __restrict__ menc) {
  int i = blockIdx.x * blockDim.x + threadIdx.x;
  if (i < N_NODES * DCH) agg[i] = 0.0f;
  if (i < N_NODES) { denom[i] = 0.0f; menc[i] = 0x007FFFFFu; } // enc(-inf)
}

// ---- fused 4-matrix GEMM: [16x64] @ [64x64] per wave, WMMA f16->f32 ----
// block = 128 threads = 4 waves; wave w computes q/k/v/s for one 16-row tile.
__global__ void __launch_bounds__(128) GtcGemm4_kernel(
    const _Float16* __restrict__ h16, const _Float16* __restrict__ wsw,
    const float* __restrict__ bq, const float* __restrict__ bk,
    const float* __restrict__ bv, const float* __restrict__ bs,
    float* __restrict__ q, float* __restrict__ k,
    float* __restrict__ v, float* __restrict__ s) {
  const int lane = threadIdx.x & 31;
  const int mat  = threadIdx.x >> 5;      // wave-uniform: 0..3 -> q,k,v,s
  const int tile = blockIdx.x;            // 16-row node tile (N=50000=3125*16)
  const int row  = lane & 15;
  const int hi   = lane >> 4;

  const float* bias;
  float* outp;
  switch (mat) {
    case 0:  bias = bq; outp = q; break;
    case 1:  bias = bk; outp = k; break;
    case 2:  bias = bv; outp = v; break;
    default: bias = bs; outp = s; break;
  }
  const _Float16* wbase = wsw + mat * (4 * 2 * 32 * 16);

  // A as loaded here (contiguous k, same order as B): lanes 0-15 row=lane, K = 0..15; lanes 16-31 K = 16..31
  const _Float16* arow = h16 + (size_t)(tile * 16 + row) * 64;
  const int kb = hi * 16;
  v8h a0lo = *(const v8h*)(arow + kb);
  v8h a0hi = *(const v8h*)(arow + kb + 8);
  v8h a1lo = *(const v8h*)(arow + 32 + kb);
  v8h a1hi = *(const v8h*)(arow + 32 + kb + 8);
  v16h a0, a1;
#pragma unroll
  for (int j = 0; j < 8; ++j) {
    a0[j] = a0lo[j]; a0[j + 8] = a0hi[j];
    a1[j] = a1lo[j]; a1[j + 8] = a1hi[j];
  }

#pragma unroll
  for (int t = 0; t < 4; ++t) {
    const int n = t * 16 + row;
    const float bval = bias[n];
    v8f c;
#pragma unroll
    for (int i = 0; i < 8; ++i) c[i] = bval;   // bias broadcast into accumulator
    v16h b0 = *(const v16h*)(wbase + ((t * 2 + 0) * 32 + lane) * 16);
    v16h b1 = *(const v16h*)(wbase + ((t * 2 + 1) * 32 + lane) * 16);
    c = __builtin_amdgcn_wmma_f32_16x16x32_f16(false, a0, false, b0, (short)0, c, false, false);
    c = __builtin_amdgcn_wmma_f32_16x16x32_f16(false, a1, false, b1, (short)0, c, false, false);
    float* orow = outp + (size_t)(tile * 16 + hi * 8) * 64 + n;
#pragma unroll
    for (int i = 0; i < 8; ++i) orow[i * 64] = c[i];  // C layout: VGPR i -> row i+8*hi
  }
}

// ---- per-edge attention logits + segment max (16 lanes per edge) ----
__global__ void GtcScores_kernel(const long long* __restrict__ ei,
                                 const float* __restrict__ q, const float* __restrict__ k,
                                 float* __restrict__ scores, unsigned* __restrict__ menc) {
  int gtid = blockIdx.x * blockDim.x + threadIdx.x;   // E*16 threads exactly
  int e  = gtid >> 4;
  int c4 = gtid & 15;
  int src = (int)ei[e];
  int dst = (int)ei[N_EDGES + e];
  const float4 qv = *(const float4*)(q + (size_t)dst * 64 + c4 * 4);
  const float4 kv = *(const float4*)(k + (size_t)src * 64 + c4 * 4);
  float p = qv.x * kv.x + qv.y * kv.y + qv.z * kv.z + qv.w * kv.w;
  p += __shfl_xor(p, 8);
  p += __shfl_xor(p, 4);
  p += __shfl_xor(p, 2);
  p += __shfl_xor(p, 1);
  if (c4 == 0) {
    float sc = p * 0.125f;   // 1/sqrt(64)
    scores[e] = sc;
    atomicMax(menc + dst, enc_f32(sc));
  }
}

// ---- exp(score - m[dst]) and denominator accumulation ----
__global__ void GtcExDen_kernel(const long long* __restrict__ ei,
                                const float* __restrict__ scores,
                                const unsigned* __restrict__ menc,
                                float* __restrict__ exv, float* __restrict__ denom) {
  int e = blockIdx.x * blockDim.x + threadIdx.x;
  if (e >= N_EDGES) return;
  int dst = (int)ei[N_EDGES + e];
  float ex = __expf(scores[e] - dec_f32(menc[dst]));
  exv[e] = ex;
  unsafeAtomicAdd(denom + dst, ex);
}

// ---- agg[dst] += ex * v[src]  (16 lanes per edge, 4 channels each) ----
__global__ void GtcAgg_kernel(const long long* __restrict__ ei,
                              const float* __restrict__ exv,
                              const float* __restrict__ v, float* __restrict__ agg) {
  int gtid = blockIdx.x * blockDim.x + threadIdx.x;   // E*16 threads exactly
  int e  = gtid >> 4;
  int c4 = gtid & 15;
  int src = (int)ei[e];
  int dst = (int)ei[N_EDGES + e];
  float w = exv[e];
  const float4 vv = *(const float4*)(v + (size_t)src * 64 + c4 * 4);
  float* ap = agg + (size_t)dst * 64 + c4 * 4;
  unsafeAtomicAdd(ap + 0, w * vv.x);
  unsafeAtomicAdd(ap + 1, w * vv.y);
  unsafeAtomicAdd(ap + 2, w * vv.z);
  unsafeAtomicAdd(ap + 3, w * vv.w);
}

// ---- finalize: agg/denom + skip, then layer epilogue ----
// mode 0: silu(t)          (layer 1)
// mode 1: silu(t + extra)  (layer 2 residual)
// mode 2: t + 0.1*extra    (layer 3 + noise)
__global__ void GtcFinal_kernel(const float* __restrict__ agg, const float* __restrict__ denom,
                                const float* __restrict__ s, const float* __restrict__ extra,
                                float* __restrict__ out, int mode) {
  int i = blockIdx.x * blockDim.x + threadIdx.x;
  if (i >= N_NODES * DCH) return;
  int node = i >> 6;
  float de = denom[node];
  float t = agg[i] / (de > 0.0f ? de : 1.0f) + s[i];
  if (mode == 0) {
    out[i] = t / (1.0f + __expf(-t));
  } else if (mode == 1) {
    float u = t + extra[i];
    out[i] = u / (1.0f + __expf(-u));
  } else {
    out[i] = t + 0.1f * extra[i];
  }
}

extern "C" void kernel_launch(void* const* d_in, const int* in_sizes, int n_in,
                              void* d_out, int out_size, void* d_ws, size_t ws_size,
                              hipStream_t stream) {
  const float*     x  = (const float*)d_in[0];
  const long long* ei = (const long long*)d_in[1];   // int64 [2, E]
  const float*     z  = (const float*)d_in[2];

  const float* W[3][4];
  const float* B[3][4];
  for (int l = 0; l < 3; ++l)
    for (int m = 0; m < 4; ++m) {
      W[l][m] = (const float*)d_in[3 + l * 8 + m * 2];
      B[l][m] = (const float*)d_in[3 + l * 8 + m * 2 + 1];
    }

  // ---- workspace bump allocator (256B aligned) ----
  char* ws = (char*)d_ws;
  size_t off = 0;
  auto alloc = [&](size_t bytes) -> void* {
    void* p = ws + off;
    off = (off + bytes + 255) & ~(size_t)255;
    return p;
  };
  const size_t WSWL = 4 * 2 * 32 * 16;                       // f16 elems per matrix, used as layer stride
  _Float16* wsw  = (_Float16*)alloc(3 * WSWL * sizeof(_Float16));
  _Float16* h16  = (_Float16*)alloc((size_t)N_NODES * DCH * sizeof(_Float16));
  float* qb   = (float*)alloc((size_t)N_NODES * DCH * sizeof(float));
  float* kb   = (float*)alloc((size_t)N_NODES * DCH * sizeof(float));
  float* vb   = (float*)alloc((size_t)N_NODES * DCH * sizeof(float));
  float* sb   = (float*)alloc((size_t)N_NODES * DCH * sizeof(float));
  float* agg1 = (float*)alloc((size_t)N_NODES * DCH * sizeof(float)); // -> h1
  float* agg2 = (float*)alloc((size_t)N_NODES * DCH * sizeof(float)); // -> h2
  float* ebuf = (float*)alloc((size_t)N_EDGES * sizeof(float));       // scores, then ex
  float* denom = (float*)alloc((size_t)N_NODES * sizeof(float));
  unsigned* menc = (unsigned*)alloc((size_t)N_NODES * sizeof(unsigned));

  // pre-swizzle all 12 weight matrices to WMMA B layout (f16)
  for (int l = 0; l < 3; ++l)
    GtcWconv_kernel<<<(4 * 64 * 64 + 255) / 256, 256, 0, stream>>>(
        W[l][0], W[l][1], W[l][2], W[l][3], wsw + l * WSWL);

  const int NT   = N_NODES * DCH;
  const int nblk = (NT + 255) / 256;
  const float* hin[3]  = { x, agg1, agg2 };        // layer inputs
  float*       aggb[3] = { agg1, agg2, agg1 };     // scatter targets (agg1 reused in L3)
  float*       outb[3] = { agg1, agg2, (float*)d_out }; // finalize in place -> next h
  const float* extra[3] = { nullptr, agg1, z };    // L2 residual, L3 noise

  for (int l = 0; l < 3; ++l) {
    GtcF32toF16_kernel<<<nblk, 256, 0, stream>>>(hin[l], h16, NT);
    GtcInit_kernel<<<nblk, 256, 0, stream>>>(aggb[l], denom, menc);
    GtcGemm4_kernel<<<N_NODES / 16, 128, 0, stream>>>(
        h16, wsw + l * WSWL, B[l][0], B[l][1], B[l][2], B[l][3], qb, kb, vb, sb);
    GtcScores_kernel<<<(N_EDGES * 16) / 256, 256, 0, stream>>>(ei, qb, kb, ebuf, menc);
    GtcExDen_kernel<<<(N_EDGES + 255) / 256, 256, 0, stream>>>(ei, ebuf, menc, ebuf, denom);
    GtcAgg_kernel<<<(N_EDGES * 16) / 256, 256, 0, stream>>>(ei, ebuf, vb, aggb[l]);
    GtcFinal_kernel<<<nblk, 256, 0, stream>>>(aggb[l], denom, sb, extra[l], outb[l], l);
  }
}
